// TransFusionHead_28673201668538
// MI455X (gfx1250) — compile-verified, matrix operand order checked by reading, not run on hardware
//
#include <hip/hip_runtime.h>

typedef _Float16 v16h __attribute__((ext_vector_type(16)));
typedef _Float16 v8h  __attribute__((ext_vector_type(8)));
typedef float    v8f  __attribute__((ext_vector_type(8)));

#define GG    128
#define HW    16384
#define HIDC  128
#define CINC  384
#define NPROPC 200
#define NPP   224          // proposals padded to multiple of 32
#define NCLSC 10
#define NCHUNK 4096        // spatial chunk for im2col (multiple of 32)

#define F_ST16 1
#define F_NMAJ 2
#define F_RELU 4

// ---------------------------------------------------------------- WMMA utils
__device__ __forceinline__ v8f wmma16(v16h a, v16h b, v8f c) {
  return __builtin_amdgcn_wmma_f32_16x16x32_f16(false, a, false, b, (short)0, c, false, false);
}

// A/B fragment from [rows][K] K-contiguous f16, per-lane pointer already offset
// by row*K + kbase + ((lane>>4)&1)*8. Two b128 loads (global or ds).
__device__ __forceinline__ v16h load_frag(const _Float16* p) {
  v8h lo = *(const v8h*)(p);
  v8h hi = *(const v8h*)(p + 16);
  v16h r;
#pragma unroll
  for (int i = 0; i < 8; ++i) { r[i] = lo[i]; r[i + 8] = hi[i]; }
  return r;
}

// CDNA5 async global->LDS copy (ASYNCcnt), 16B per lane. GV addressing mode.
__device__ __forceinline__ void async_copy16(unsigned ldsoff, const void* gaddr) {
  asm volatile("global_load_async_to_lds_b128 %0, %1, off"
               :: "v"(ldsoff), "v"(gaddr) : "memory");
}
__device__ __forceinline__ void async_wait0() {
  asm volatile("s_wait_asynccnt 0" ::: "memory");
}

__device__ __forceinline__ void store_tile(v8f acc, int m0, int n0, int lr, int hi,
                                           const float* bias, void* D, int ldD,
                                           int mValid, int nValid, size_t bofs, int flags) {
  int n = n0 + lr;
  if (n >= nValid) return;
  int mb = m0 + 8 * hi;
  for (int r = 0; r < 8; ++r) {
    int m = mb + r;
    if (m >= mValid) continue;
    float v = acc[r] + bias[m];
    if (flags & F_RELU) v = v > 0.f ? v : 0.f;
    size_t off = bofs + ((flags & F_NMAJ) ? ((size_t)n * ldD + m) : ((size_t)m * ldD + n));
    if (flags & F_ST16) ((_Float16*)D)[off] = (_Float16)v;
    else                ((float*)D)[off] = v;
  }
}

// ---------------------------------------------------------------------------
// Conv implicit-GEMM: D[m][n] = sum_k A[m][k]*B[n][k] + bias[m] (m-major f32).
// Block = 4 waves = 128x32 output strip; the shared 32(n)x32(k) B tile is
// staged in LDS via async-to-LDS, double buffered, overlapping the WMMAs.
// grid = (N/32), block = 128.
__global__ void __launch_bounds__(128) gemm_conv_k(
    const _Float16* __restrict__ A, const _Float16* __restrict__ B,
    const float* __restrict__ bias, float* __restrict__ D,
    int M, int K, int ldD, int mValid, int nValid) {
  __shared__ _Float16 Bt[2][32][32];
  int tid = threadIdx.x;
  int lane = tid & 31;
  int w = tid >> 5;               // wave id 0..3 -> m-tile pair
  int n0 = blockIdx.x * 32;
  int m0 = w * 32;
  int lr = lane & 15;
  int hi = (lane >> 4) & 1;
  int koff = hi * 8;
  // per-thread async slice: 128 thr x 16B = 2048B = full 32x32 f16 tile
  int prow = tid >> 2;            // 0..31 (n-local row)
  int pchk = (tid & 3) * 8;       // k-local halves
  unsigned lds0 = (unsigned)(size_t)(&Bt[0][0][0]) + (unsigned)(prow * 64 + (tid & 3) * 16);
  unsigned lds1 = (unsigned)(size_t)(&Bt[1][0][0]) + (unsigned)(prow * 64 + (tid & 3) * 16);
  const _Float16* gsrc = B + (size_t)(n0 + prow) * K + pchk;

  bool act = (m0 < M);
  const _Float16* pa0 = A + (size_t)(m0 + lr) * K + koff;
  const _Float16* pa1 = pa0 + (size_t)16 * K;
  v8f a00 = {}, a01 = {}, a10 = {}, a11 = {};

  async_copy16(lds0, gsrc);       // prologue: tile k=0 -> buf 0
  async_wait0();
  __syncthreads();

  int cb = 0;
  for (int k = 0; k < K; k += 32) {
    if (k + 32 < K) async_copy16(cb ? lds0 : lds1, gsrc + (k + 32));
    if (act) {
      v16h fa0 = load_frag(pa0 + k);
      v16h fa1 = load_frag(pa1 + k);
      v16h fb0 = load_frag(&Bt[cb][lr][0] + koff);
      v16h fb1 = load_frag(&Bt[cb][16 + lr][0] + koff);
      a00 = wmma16(fa0, fb0, a00);
      a01 = wmma16(fa0, fb1, a01);
      a10 = wmma16(fa1, fb0, a10);
      a11 = wmma16(fa1, fb1, a11);
    }
    async_wait0();
    __syncthreads();
    cb ^= 1;
  }
  if (act) {
    store_tile(a00, m0,      n0,      lr, hi, bias, D, ldD, mValid, nValid, 0, 0);
    store_tile(a01, m0,      n0 + 16, lr, hi, bias, D, ldD, mValid, nValid, 0, 0);
    store_tile(a10, m0 + 16, n0,      lr, hi, bias, D, ldD, mValid, nValid, 0, 0);
    store_tile(a11, m0 + 16, n0 + 16, lr, hi, bias, D, ldD, mValid, nValid, 0, 0);
  }
}

// Generic (small) GEMM: one wave per 32x32 tile. grid=(N/32, M/32, batch).
__global__ void gemm_wmma_k(const _Float16* __restrict__ A, const _Float16* __restrict__ B,
                            const float* __restrict__ bias, void* __restrict__ D,
                            int K, int ldD, int mValid, int nValid,
                            long long sA, long long sB, long long sD, int flags) {
  int lane = threadIdx.x;
  int bz = blockIdx.z;
  const _Float16* Ab = A + (size_t)bz * (size_t)sA;
  const _Float16* Bb = B + (size_t)bz * (size_t)sB;
  int m0 = blockIdx.y * 32, n0 = blockIdx.x * 32;
  int lr = lane & 15;
  int hi = (lane >> 4) & 1;
  int koff = hi * 8;
  const _Float16* pa0 = Ab + (size_t)(m0 + lr) * K + koff;
  const _Float16* pa1 = pa0 + (size_t)16 * K;
  const _Float16* pb0 = Bb + (size_t)(n0 + lr) * K + koff;
  const _Float16* pb1 = pb0 + (size_t)16 * K;
  v8f a00 = {}, a01 = {}, a10 = {}, a11 = {};
  for (int k = 0; k < K; k += 32) {
    v16h fa0 = load_frag(pa0 + k);
    v16h fa1 = load_frag(pa1 + k);
    v16h fb0 = load_frag(pb0 + k);
    v16h fb1 = load_frag(pb1 + k);
    a00 = wmma16(fa0, fb0, a00);
    a01 = wmma16(fa0, fb1, a01);
    a10 = wmma16(fa1, fb0, a10);
    a11 = wmma16(fa1, fb1, a11);
  }
  size_t bofs = (size_t)bz * (size_t)sD;
  store_tile(a00, m0,      n0,      lr, hi, bias, D, ldD, mValid, nValid, bofs, flags);
  store_tile(a01, m0,      n0 + 16, lr, hi, bias, D, ldD, mValid, nValid, bofs, flags);
  store_tile(a10, m0 + 16, n0,      lr, hi, bias, D, ldD, mValid, nValid, bofs, flags);
  store_tile(a11, m0 + 16, n0 + 16, lr, hi, bias, D, ldD, mValid, nValid, bofs, flags);
}

// --------------------------------------------------- flash attention (Dh=16)
// Q,K: [B][*][128] token-major f16; Vt: [B][128][NK] channel-major f16.
// grid=(NPP/16,1,B), block=256 (8 waves = 8 heads). Online softmax, WMMA for
// S=Q*K^T (Dh padded 16->32 with zeros) and O += P*V.
__global__ void flash_attn_k(const _Float16* __restrict__ Q, const _Float16* __restrict__ Kh,
                             const _Float16* __restrict__ Vt, _Float16* __restrict__ O,
                             int NK, int kValid,
                             long long sQ, long long sK, long long sV, long long sO) {
  __shared__ float S[8][16][32];
  __shared__ float Mr[8][16], Lr[8][16], Ar[8][16];
  int lane = threadIdx.x & 31;
  int h = threadIdx.x >> 5;
  int b = blockIdx.z;
  int q0 = blockIdx.x * 16;
  const _Float16* Qb = Q  + (size_t)b * sQ;
  const _Float16* Kb = Kh + (size_t)b * sK;
  const _Float16* Vb = Vt + (size_t)b * sV;
  _Float16*       Ob = O  + (size_t)b * sO;
  int hc = h * 16;
  int lr = lane & 15;
  int hi = (lane >> 4) & 1;
  int koff = hi * 8;
  v8h qv = *(const v8h*)(Qb + (size_t)(q0 + lr) * HIDC + hc + koff);
  v16h qf;
#pragma unroll
  for (int i = 0; i < 8; ++i) { qf[i] = qv[i]; qf[i + 8] = (_Float16)0.f; }
  v8f acc = {};
  if (lane < 16) { Mr[h][lane] = -1e30f; Lr[h][lane] = 0.f; }
  __syncthreads();
  for (int kc = 0; kc < NK; kc += 32) {
    v8h kv0 = *(const v8h*)(Kb + (size_t)(kc + lr) * HIDC + hc + koff);
    v8h kv1 = *(const v8h*)(Kb + (size_t)(kc + 16 + lr) * HIDC + hc + koff);
    v16h bf0, bf1;
#pragma unroll
    for (int i = 0; i < 8; ++i) {
      bf0[i] = kv0[i]; bf0[i + 8] = (_Float16)0.f;
      bf1[i] = kv1[i]; bf1[i + 8] = (_Float16)0.f;
    }
    v8f s0 = {}, s1 = {};
    s0 = wmma16(qf, bf0, s0);
    s1 = wmma16(qf, bf1, s1);
    int qrow = 8 * hi;
#pragma unroll
    for (int r = 0; r < 8; ++r) {
      S[h][qrow + r][lr]      = (kc + lr      < kValid) ? s0[r] * 0.25f : -1e30f;
      S[h][qrow + r][16 + lr] = (kc + 16 + lr < kValid) ? s1[r] * 0.25f : -1e30f;
    }
    __syncthreads();
    if (lane < 16) {
      int q = lane;
      float mold = Mr[h][q], mc = mold;
      for (int j = 0; j < 32; ++j) mc = fmaxf(mc, S[h][q][j]);
      float alpha = __expf(mold - mc);
      float ls = Lr[h][q] * alpha;
      for (int j = 0; j < 32; ++j) { float pv = __expf(S[h][q][j] - mc); S[h][q][j] = pv; ls += pv; }
      Mr[h][q] = mc; Lr[h][q] = ls; Ar[h][q] = alpha;
    }
    __syncthreads();
    v16h pf;
#pragma unroll
    for (int i = 0; i < 8; ++i) {
      pf[i]     = (_Float16)S[h][lr][koff + i];
      pf[i + 8] = (_Float16)S[h][lr][koff + 16 + i];
    }
    const _Float16* vp = Vb + (size_t)(hc + lr) * NK + kc + koff;
    v8h vv0 = *(const v8h*)(vp);
    v8h vv1 = *(const v8h*)(vp + 16);
    v16h vf;
#pragma unroll
    for (int i = 0; i < 8; ++i) { vf[i] = vv0[i]; vf[i + 8] = vv1[i]; }
#pragma unroll
    for (int r = 0; r < 8; ++r) acc[r] *= Ar[h][8 * hi + r];
    acc = wmma16(pf, vf, acc);
    __syncthreads();
  }
#pragma unroll
  for (int r = 0; r < 8; ++r) {
    int q = 8 * hi + r;
    Ob[(size_t)(q0 + q) * HIDC + hc + lr] = (_Float16)(acc[r] / Lr[h][q]);
  }
}

// --------------------------------------------------------- elementwise utils
__global__ void cvt_f16(const float* __restrict__ s, _Float16* __restrict__ d, int nsrc, int ntot) {
  int i = blockIdx.x * blockDim.x + threadIdx.x;
  if (i < ntot) d[i] = (i < nsrc) ? (_Float16)s[i] : (_Float16)0.f;
}

// im2col (3x3, pad 1) of [C][G][G] f32 -> col [NCHUNK][C*9] f16, k=(cin,dy,dx)
__global__ void im2col3x3(const float* __restrict__ in, _Float16* __restrict__ col, int C, int n0) {
  long long K = (long long)C * 9;
  long long total = (long long)NCHUNK * K;
  long long i = (long long)blockIdx.x * blockDim.x + threadIdx.x;
  if (i >= total) return;
  int k = (int)(i % K);
  int nl = (int)(i / K);
  int n = n0 + nl;
  int cin = k / 9, r = k % 9;
  int y = n / GG + r / 3 - 1;
  int x = n % GG + r % 3 - 1;
  float v = 0.f;
  if (y >= 0 && y < GG && x >= 0 && x < GG) v = in[(size_t)cin * HW + (size_t)y * GG + x];
  col[(size_t)nl * K + k] = (_Float16)v;
}

__global__ void bnrelu_ch(float* __restrict__ x, const float* m, const float* v,
                          const float* g, const float* bb, int Nsp, long long total) {
  long long i = (long long)blockIdx.x * blockDim.x + threadIdx.x;
  if (i >= total) return;
  int c = (int)(i / Nsp);
  float val = (x[i] - m[c]) * (g[c] * rsqrtf(v[c] + 1e-5f)) + bb[c];
  x[i] = fmaxf(val, 0.f);
}

// sigmoid + 3x3 local-max NMS (classes 8,9 pass through; borders suppressed)
__global__ void nms_k(const float* __restrict__ dh, float* __restrict__ out) {
  int i = blockIdx.x * blockDim.x + threadIdx.x;
  if (i >= NCLSC * HW) return;
  int c = i / HW, n = i % HW, y = n / GG, x = n % GG;
  float s = 1.f / (1.f + __expf(-dh[i]));
  float keep;
  if (c >= 8) keep = s;
  else if (y == 0 || y == GG - 1 || x == 0 || x == GG - 1) keep = 0.f;
  else {
    float mx = -1e30f;
    for (int dy = -1; dy <= 1; ++dy)
      for (int dx = -1; dx <= 1; ++dx) {
        float q = 1.f / (1.f + __expf(-dh[c * HW + (y + dy) * GG + x + dx]));
        mx = fmaxf(mx, q);
      }
    keep = (s == mx) ? s : 0.f;
  }
  out[i] = keep;
}

// stable top-200 of [NCLS*HW] (ties -> smaller flat index). One block, 1024 thr.
__global__ void topk_k(const float* __restrict__ hm, float* __restrict__ sc,
                       int* __restrict__ cls, int* __restrict__ idx) {
  __shared__ float sv[1024];
  __shared__ int   si[1024];
  int t = threadIdx.x;
  for (int i = t; i < NCLSC * HW; i += 1024) sc[i] = hm[i];
  __syncthreads();
  for (int p = 0; p < NPROPC; ++p) {
    float bv = -1e30f; int bi = 0x7fffffff;
    for (int i = t; i < NCLSC * HW; i += 1024) {
      float v = sc[i];
      if (v > bv) { bv = v; bi = i; }
    }
    sv[t] = bv; si[t] = bi;
    __syncthreads();
    for (int s = 512; s > 0; s >>= 1) {
      if (t < s) {
        if (sv[t + s] > sv[t] || (sv[t + s] == sv[t] && si[t + s] < si[t])) {
          sv[t] = sv[t + s]; si[t] = si[t + s];
        }
      }
      __syncthreads();
    }
    if (t == 0) { int bb = si[0]; cls[p] = bb / HW; idx[p] = bb % HW; sc[bb] = -1e30f; }
    __syncthreads();
  }
}

// query[b][n][c] = lidar[b][c][idx]+cls_w[c][cls]+cls_b[c]; pads rows>=200 to 0
__global__ void gather_query(const float* __restrict__ lidar, const float* __restrict__ clsw,
                             const float* __restrict__ clsb, const float* __restrict__ bev,
                             const int* __restrict__ cls, const int* __restrict__ idx,
                             float* __restrict__ query, float* __restrict__ qpos) {
  int n = blockIdx.x, b = blockIdx.y, c = threadIdx.x;
  float v = 0.f;
  if (n < NPROPC) {
    int id = idx[b * NPROPC + n], cl = cls[b * NPROPC + n];
    v = lidar[((size_t)b * HIDC + c) * HW + id] + clsw[c * NCLSC + cl] + clsb[c];
    if (c < 2) qpos[((size_t)b * NPP + n) * 2 + c] = bev[((size_t)b * HW + id) * 2 + c];
  } else if (c < 2) {
    qpos[((size_t)b * NPP + n) * 2 + c] = 0.f;
  }
  query[((size_t)b * NPP + n) * HIDC + c] = v;
}

// pos-embed layer1: relu(bn1d(xy @ w1^T + b1)) -> f16 [B][N][128]
__global__ void posembed1(const float* __restrict__ xy, const float* __restrict__ w1,
                          const float* __restrict__ b1, const float* __restrict__ bnb,
                          const float* __restrict__ bng, const float* __restrict__ bnm,
                          const float* __restrict__ bnv, _Float16* __restrict__ out, int N) {
  int n = blockIdx.x, b = blockIdx.y, c = threadIdx.x;
  const float* p = xy + ((size_t)b * N + n) * 2;
  float h = p[0] * w1[c * 2] + p[1] * w1[c * 2 + 1] + b1[c];
  h = (h - bnm[c]) * (bng[c] * rsqrtf(bnv[c] + 1e-5f)) + bnb[c];
  out[((size_t)b * N + n) * HIDC + c] = (_Float16)fmaxf(h, 0.f);
}

__global__ void add_f16(const float* __restrict__ a, const float* __restrict__ b,
                        _Float16* __restrict__ o, long long n) {
  long long i = (long long)blockIdx.x * blockDim.x + threadIdx.x;
  if (i < n) o[i] = (_Float16)(a[i] + b[i]);
}

// kv[b][n][c] = lidar[b][c][n] + kpe[b][n][c]
__global__ void kv_add(const float* __restrict__ lidar, const float* __restrict__ kpe,
                       _Float16* __restrict__ kv) {
  int n = blockIdx.x, b = blockIdx.y, c = threadIdx.x;
  kv[((size_t)b * HW + n) * HIDC + c] =
      (_Float16)(lidar[((size_t)b * HIDC + c) * HW + n] + kpe[((size_t)b * HW + n) * HIDC + c]);
}

// out = layernorm(x + r) * g + b ; writes f32 and f16. grid=(NPP,B), block=128
__global__ void ln_k(const float* __restrict__ x, const float* __restrict__ r,
                     const float* __restrict__ g, const float* __restrict__ bb,
                     float* __restrict__ o32, _Float16* __restrict__ o16) {
  __shared__ float red[HIDC];
  __shared__ float stat[2];
  int n = blockIdx.x, b = blockIdx.y, c = threadIdx.x;
  size_t base = ((size_t)b * NPP + n) * HIDC;
  float v = x[base + c] + r[base + c];
  red[c] = v;
  __syncthreads();
  for (int s = 64; s > 0; s >>= 1) { if (c < s) red[c] += red[c + s]; __syncthreads(); }
  if (c == 0) stat[0] = red[0] * (1.f / HIDC);
  __syncthreads();
  float d = v - stat[0];
  red[c] = d * d;
  __syncthreads();
  for (int s = 64; s > 0; s >>= 1) { if (c < s) red[c] += red[c + s]; __syncthreads(); }
  if (c == 0) stat[1] = red[0] * (1.f / HIDC);
  __syncthreads();
  float y = d * rsqrtf(stat[1] + 1e-5f) * g[c] + bb[c];
  o32[base + c] = y;
  o16[base + c] = (_Float16)y;
}

__global__ void bnrelu_head(const float* __restrict__ x, _Float16* __restrict__ o,
                            const float* m, const float* v, const float* g,
                            const float* bb, long long total) {
  long long i = (long long)blockIdx.x * blockDim.x + threadIdx.x;
  if (i >= total) return;
  int c = (int)(i & 63);
  float val = (x[i] - m[c]) * (g[c] * rsqrtf(v[c] + 1e-5f)) + bb[c];
  o[i] = (_Float16)fmaxf(val, 0.f);
}

__global__ void center_add(float* __restrict__ out, const float* __restrict__ qpos) {
  int i = blockIdx.x * blockDim.x + threadIdx.x;
  if (i >= 2 * 2 * NPROPC) return;
  int b = i / (2 * NPROPC), rm = i % (2 * NPROPC), a = rm / NPROPC, n = rm % NPROPC;
  out[i] += qpos[((size_t)b * NPP + n) * 2 + a];
}

__global__ void qh_gather(const float* __restrict__ hm, const int* __restrict__ idx,
                          float* __restrict__ out) {
  int i = blockIdx.x * blockDim.x + threadIdx.x;
  if (i >= 2 * NCLSC * NPROPC) return;
  int b = i / (NCLSC * NPROPC), rm = i % (NCLSC * NPROPC), c = rm / NPROPC, p = rm % NPROPC;
  out[i] = hm[(size_t)b * NCLSC * HW + (size_t)c * HW + idx[b * NPROPC + p]];
}

// ------------------------------------------------------------------- driver
extern "C" void kernel_launch(void* const* d_in, const int* in_sizes, int n_in,
                              void* d_out, int out_size, void* d_ws, size_t ws_size,
                              hipStream_t stream) {
  (void)in_sizes; (void)n_in; (void)out_size; (void)ws_size;
  // param indices (jax tree-flatten, alphabetical dict keys); 0=inputs, 103=bev_pos
  const float* inputs = (const float*)d_in[0];
  auto P = [&](int i) { return (const float*)d_in[i]; };
  const float* bev = P(103);
  float* out = (float*)d_out;

  char* ws = (char*)d_ws;
  size_t cur = 0;
  auto alloc = [&](size_t bytes) -> char* {
    char* p = ws + cur;
    cur += (bytes + 255) & ~(size_t)255;
    return p;
  };
  _Float16* wsh16 = (_Float16*)alloc((size_t)128 * 3456 * 2);
  _Float16* whm1  = (_Float16*)alloc((size_t)128 * 1152 * 2);
  _Float16* whm2  = (_Float16*)alloc((size_t)32 * 1152 * 2);
  _Float16* wat[8];
  for (int i = 0; i < 8; ++i) wat[i] = (_Float16*)alloc((size_t)128 * 128 * 2);
  _Float16* wps2 = (_Float16*)alloc((size_t)128 * 128 * 2);
  _Float16* wpc2 = (_Float16*)alloc((size_t)128 * 128 * 2);
  _Float16* wff1 = (_Float16*)alloc((size_t)256 * 128 * 2);
  _Float16* wff2 = (_Float16*)alloc((size_t)128 * 256 * 2);
  _Float16* w1buf = (_Float16*)alloc((size_t)64 * 128 * 2);
  _Float16* w2buf = (_Float16*)alloc((size_t)32 * 64 * 2);
  _Float16* col   = (_Float16*)alloc((size_t)NCHUNK * 3456 * 2);
  float* lidar    = (float*)alloc((size_t)2 * 128 * HW * 4);
  float* xbuf     = (float*)alloc((size_t)128 * HW * 4);
  float* hmflat   = (float*)alloc((size_t)2 * NCLSC * HW * 4);
  float* tksc     = (float*)alloc((size_t)NCLSC * HW * 4);
  int* topcls     = (int*)alloc((size_t)2 * NPROPC * 4);
  int* topidx     = (int*)alloc((size_t)2 * NPROPC * 4);
  float* query32  = (float*)alloc((size_t)2 * NPP * 128 * 4);
  float* qpos     = (float*)alloc((size_t)2 * NPP * 2 * 4);
  _Float16* pehq  = (_Float16*)alloc((size_t)2 * NPP * 128 * 2);
  float* qpe32    = (float*)alloc((size_t)2 * NPP * 128 * 4);
  _Float16* pehk  = (_Float16*)alloc((size_t)2 * HW * 128 * 2);
  float* kpe32    = (float*)alloc((size_t)2 * HW * 128 * 4);
  _Float16* qsa   = (_Float16*)alloc((size_t)2 * NPP * 128 * 2);
  _Float16* qh    = (_Float16*)alloc((size_t)2 * NPP * 128 * 2);
  _Float16* kh    = (_Float16*)alloc((size_t)2 * NPP * 128 * 2);
  _Float16* vt    = (_Float16*)alloc((size_t)2 * 128 * NPP * 2);
  _Float16* o16   = (_Float16*)alloc((size_t)2 * NPP * 128 * 2);
  float* attn32   = (float*)alloc((size_t)2 * NPP * 128 * 4);
  float* q1_32    = (float*)alloc((size_t)2 * NPP * 128 * 4);
  _Float16* q1_16 = (_Float16*)alloc((size_t)2 * NPP * 128 * 2);
  _Float16* q2in  = (_Float16*)alloc((size_t)2 * NPP * 128 * 2);
  _Float16* kv16  = (_Float16*)alloc((size_t)2 * HW * 128 * 2);
  _Float16* qh2   = (_Float16*)alloc((size_t)2 * NPP * 128 * 2);
  _Float16* kh2   = (_Float16*)alloc((size_t)2 * HW * 128 * 2);
  _Float16* vt2   = (_Float16*)alloc((size_t)2 * 128 * HW * 2);
  _Float16* o2    = (_Float16*)alloc((size_t)2 * NPP * 128 * 2);
  float* attn2_32 = (float*)alloc((size_t)2 * NPP * 128 * 4);
  float* q2_32    = (float*)alloc((size_t)2 * NPP * 128 * 4);
  _Float16* q2_16 = (_Float16*)alloc((size_t)2 * NPP * 128 * 2);
  _Float16* ffh   = (_Float16*)alloc((size_t)2 * NPP * 256 * 2);
  float* ff32     = (float*)alloc((size_t)2 * NPP * 128 * 4);
  float* qF32     = (float*)alloc((size_t)2 * NPP * 128 * 4);
  _Float16* qF16  = (_Float16*)alloc((size_t)2 * NPP * 128 * 2);
  float* hh32     = (float*)alloc((size_t)2 * NPP * 64 * 4);
  _Float16* hh16  = (_Float16*)alloc((size_t)2 * NPP * 64 * 2);

  auto cvt = [&](int pidx, _Float16* dst, int nsrc, int ntot) {
    cvt_f16<<<dim3((ntot + 255) / 256), 256, 0, stream>>>(P(pidx), dst, nsrc, ntot);
  };
  auto gemm = [&](const _Float16* A, const _Float16* B, const float* bias, void* D,
                  int M, int N, int K, int ldD, int mValid, int nValid,
                  long long sA, long long sB, long long sD, int nb, int flags) {
    dim3 g(N / 32, M / 32, nb);
    gemm_wmma_k<<<g, 32, 0, stream>>>(A, B, bias, D, K, ldD, mValid, nValid, sA, sB, sD, flags);
  };
  auto gemmc = [&](const _Float16* A, const _Float16* B, const float* bias, float* D,
                   int M, int N, int K, int ldD, int mValid, int nValid) {
    gemm_conv_k<<<dim3(N / 32), 128, 0, stream>>>(A, B, bias, D, M, K, ldD, mValid, nValid);
  };

  // ---- weight conversion (f32 -> f16, zero-padded where M < 32) ----
  cvt(102, wsh16, 128 * 3456, 128 * 3456);
  cvt(98,  whm1,  128 * 1152, 128 * 1152);
  cvt(100, whm2,  10 * 1152,  32 * 1152);
  cvt(33, wat[0], 16384, 16384);  // self wq
  cvt(31, wat[1], 16384, 16384);  // self wk
  cvt(34, wat[2], 16384, 16384);  // self wv
  cvt(32, wat[3], 16384, 16384);  // self wo
  cvt(11, wat[4], 16384, 16384);  // cross wq
  cvt(9,  wat[5], 16384, 16384);  // cross wk
  cvt(12, wat[6], 16384, 16384);  // cross wv
  cvt(10, wat[7], 16384, 16384);  // cross wo
  cvt(42, wps2, 16384, 16384);
  cvt(20, wpc2, 16384, 16384);
  cvt(43, wff1, 256 * 128, 256 * 128);
  cvt(44, wff2, 128 * 256, 128 * 256);

  // ---- conv stack per batch (implicit GEMM via chunked im2col) ----
  for (int b = 0; b < 2; ++b) {
    const float* inb = inputs + (size_t)b * CINC * HW;
    float* lidb = lidar + (size_t)b * HIDC * HW;
    float* dhb = out + 12000 + (size_t)b * NCLSC * HW;
    for (int c0 = 0; c0 < HW; c0 += NCHUNK) {
      long long tot = (long long)NCHUNK * CINC * 9;
      im2col3x3<<<dim3((unsigned)((tot + 255) / 256)), 256, 0, stream>>>(inb, col, CINC, c0);
      gemmc(wsh16, col, P(101), lidb + c0, 128, NCHUNK, 3456, HW, 128, NCHUNK);
    }
    for (int c0 = 0; c0 < HW; c0 += NCHUNK) {
      long long tot = (long long)NCHUNK * HIDC * 9;
      im2col3x3<<<dim3((unsigned)((tot + 255) / 256)), 256, 0, stream>>>(lidb, col, HIDC, c0);
      gemmc(whm1, col, P(93), xbuf + c0, 128, NCHUNK, 1152, HW, 128, NCHUNK);
    }
    bnrelu_ch<<<dim3((128 * HW + 255) / 256), 256, 0, stream>>>(
        xbuf, P(96), P(97), P(95), P(94), HW, (long long)128 * HW);
    for (int c0 = 0; c0 < HW; c0 += NCHUNK) {
      long long tot = (long long)NCHUNK * HIDC * 9;
      im2col3x3<<<dim3((unsigned)((tot + 255) / 256)), 256, 0, stream>>>(xbuf, col, HIDC, c0);
      gemmc(whm2, col, P(99), dhb + c0, 32, NCHUNK, 1152, HW, NCLSC, NCHUNK);
    }
    nms_k<<<dim3((NCLSC * HW + 255) / 256), 256, 0, stream>>>(dhb, hmflat + (size_t)b * NCLSC * HW);
    topk_k<<<1, 1024, 0, stream>>>(hmflat + (size_t)b * NCLSC * HW, tksc,
                                   topcls + b * NPROPC, topidx + b * NPROPC);
  }

  // ---- decoder ----
  gather_query<<<dim3(NPP, 2), 128, 0, stream>>>(lidar, P(2), P(1), bev, topcls, topidx,
                                                 query32, qpos);
  posembed1<<<dim3(NPP, 2), 128, 0, stream>>>(qpos, P(41), P(35), P(37), P(38), P(39), P(40),
                                              pehq, NPP);
  gemm(wps2, pehq, P(36), qpe32, 128, NPP, 128, 128, 128, NPP,
       0, (long long)NPP * 128, (long long)NPP * 128, 2, F_NMAJ);
  posembed1<<<dim3(HW, 2), 128, 0, stream>>>(bev, P(19), P(13), P(15), P(16), P(17), P(18),
                                             pehk, HW);
  gemm(wpc2, pehk, P(14), kpe32, 128, HW, 128, 128, 128, HW,
       0, (long long)HW * 128, (long long)HW * 128, 2, F_NMAJ);

  long long eP = (long long)2 * NPP * 128;
  add_f16<<<dim3((unsigned)((eP + 255) / 256)), 256, 0, stream>>>(query32, qpe32, qsa, eP);

  long long sP = (long long)NPP * 128, sVt = (long long)128 * NPP;
  gemm(wat[0], qsa, P(29), qh, 128, NPP, 128, 128, 128, NPP, 0, sP, sP, 2, F_NMAJ | F_ST16);
  gemm(wat[1], qsa, P(27), kh, 128, NPP, 128, 128, 128, NPP, 0, sP, sP, 2, F_NMAJ | F_ST16);
  gemm(wat[2], qsa, P(30), vt, 128, NPP, 128, NPP, 128, NPP, 0, sP, sVt, 2, F_ST16);
  flash_attn_k<<<dim3(NPP / 16, 1, 2), 256, 0, stream>>>(qh, kh, vt, o16, NPP, NPROPC,
                                                         sP, sP, sVt, sP);
  gemm(wat[3], o16, P(28), attn32, 128, NPP, 128, 128, 128, NPP, 0, sP, sP, 2, F_NMAJ);
  ln_k<<<dim3(NPP, 2), 128, 0, stream>>>(query32, attn32, P(22), P(21), q1_32, q1_16);

  add_f16<<<dim3((unsigned)((eP + 255) / 256)), 256, 0, stream>>>(q1_32, qpe32, q2in, eP);
  kv_add<<<dim3(HW, 2), 128, 0, stream>>>(lidar, kpe32, kv16);

  long long sK = (long long)HW * 128, sVt2 = (long long)128 * HW;
  gemm(wat[4], q2in, P(7), qh2, 128, NPP, 128, 128, 128, NPP, 0, sP, sP, 2, F_NMAJ | F_ST16);
  gemm(wat[5], kv16, P(5), kh2, 128, HW, 128, 128, 128, HW, 0, sK, sK, 2, F_NMAJ | F_ST16);
  gemm(wat[6], kv16, P(8), vt2, 128, HW, 128, HW, 128, HW, 0, sK, sVt2, 2, F_ST16);
  flash_attn_k<<<dim3(NPP / 16, 1, 2), 256, 0, stream>>>(qh2, kh2, vt2, o2, HW, HW,
                                                         sP, sK, sVt2, sP);
  gemm(wat[7], o2, P(6), attn2_32, 128, NPP, 128, 128, 128, NPP, 0, sP, sP, 2, F_NMAJ);
  ln_k<<<dim3(NPP, 2), 128, 0, stream>>>(q1_32, attn2_32, P(24), P(23), q2_32, q2_16);

  long long sF = (long long)NPP * 256;
  gemm(wff1, q2_16, P(3), ffh, 256, NPP, 128, 256, 256, NPP, 0, sP, sF, 2,
       F_NMAJ | F_ST16 | F_RELU);
  gemm(wff2, ffh, P(4), ff32, 128, NPP, 256, 128, 128, NPP, 0, sF, sP, 2, F_NMAJ);
  ln_k<<<dim3(NPP, 2), 128, 0, stream>>>(q2_32, ff32, P(26), P(25), qF32, qF16);

  // ---- prediction heads: {param base, cout, out offset (floats)} ----
  const int hb[6]   = {45, 69, 53, 77, 85, 61};   // center,height,dim,rot,vel,heatmap
  const int hcout[6] = {2, 1, 3, 2, 2, 10};
  const int hoff[6] = {0, 800, 1200, 2400, 3200, 4000};
  long long sH = (long long)NPP * 64;
  for (int i = 0; i < 6; ++i) {
    int pb = hb[i], co = hcout[i];
    cvt(pb + 6, w1buf, 64 * 128, 64 * 128);
    cvt(pb + 7, w2buf, co * 64, 32 * 64);
    gemm(w1buf, qF16, P(pb + 0), hh32, 64, NPP, 128, 64, 64, NPP, 0, sP, sH, 2, F_NMAJ);
    bnrelu_head<<<dim3((unsigned)((2 * NPP * 64 + 255) / 256)), 256, 0, stream>>>(
        hh32, hh16, P(pb + 4), P(pb + 5), P(pb + 3), P(pb + 2), (long long)2 * NPP * 64);
    gemm(w2buf, hh16, P(pb + 1), out + hoff[i], 32, NPP, 64, NPROPC, co, NPROPC,
         0, sH, (long long)co * NPROPC, 2, 0);
  }
  center_add<<<dim3((2 * 2 * NPROPC + 255) / 256), 256, 0, stream>>>(out + 0, qpos);
  qh_gather<<<dim3((2 * NCLSC * NPROPC + 255) / 256), 256, 0, stream>>>(hmflat, topidx,
                                                                        out + 8000);
}
